// MultiHeadAttention_18313740550422
// MI455X (gfx1250) — compile-verified, matrix operand order checked by reading, not run on hardware
//
#include <hip/hip_runtime.h>
#include <stdint.h>

// ---------------------------------------------------------------------------
// Types
// ---------------------------------------------------------------------------
typedef _Float16 half_t;
typedef __attribute__((ext_vector_type(16))) _Float16 v16h;
typedef __attribute__((ext_vector_type(8)))  _Float16 v8h;
typedef __attribute__((ext_vector_type(4)))  _Float16 v4h;
typedef __attribute__((ext_vector_type(8)))  float    v8f;
typedef __attribute__((ext_vector_type(4)))  unsigned int u32x4;
typedef __attribute__((ext_vector_type(8)))  int      i32x8;
typedef __attribute__((ext_vector_type(4)))  int      i32x4;

#define DEV static __device__ __forceinline__

DEV v8h ld8(const _Float16* p) { return *(const v8h*)p; }
DEV v16h cat16(v8h lo, v8h hi) {
  return __builtin_shufflevector(lo, hi, 0,1,2,3,4,5,6,7,8,9,10,11,12,13,14,15);
}
DEV v8f wmma16(v16h a, v16h b, v8f c) {
  // D(f32 16x16) = A(16x32 f16) * B(32x16 f16) + C
  return __builtin_amdgcn_wmma_f32_16x16x32_f16(false, a, false, b, (short)0, c, false, false);
}

// ---------------------------------------------------------------------------
// Tensor Data Mover (TDM) — clang-23 6-arg builtin:
//   (uint32x4 g0, int32x8 g1, int32x4 g2, int32x4 g3, int32x8, i32 cpol)
// D# group layouts per CDNA5 ISA.
// ---------------------------------------------------------------------------
#if defined(__has_builtin)
#if __has_builtin(__builtin_amdgcn_tensor_load_to_lds) && __has_builtin(__builtin_amdgcn_s_wait_tensorcnt)
#define USE_TDM 1
#endif
#endif

#ifdef USE_TDM
// dw0 bits: data_size[17:16]=1 (2B), pad_enable[20], pad_interval[24:22], pad_amount[31:25]
#define TDM_DW0_K 0x07110000  // pad every 32 DWORDs (128B row) by 4 DWORDs (-> 72-half rows)
#define TDM_DW0_V 0x06D10000  // pad every 16 DWORDs (64B row)  by 4 DWORDs (-> 40-half rows)

DEV void tdm_load(uint32_t lds_off, const void* gptr, int dw0,
                  uint32_t tdim0, uint32_t tdim1,
                  uint32_t tile0, uint32_t tile1, uint32_t stride0) {
  uint64_t ga = (uint64_t)(uintptr_t)gptr;
  u32x4 g0;
  g0[0] = 1u;                      // count=1, user descriptor
  g0[1] = lds_off;                 // LDS byte address
  g0[2] = (uint32_t)ga;            // global addr [31:0]
  g0[3] = (uint32_t)((ga >> 32) & 0x1FFFFFFu) | 0x80000000u; // addr[56:32] | type=2
  i32x8 g1;
  g1[0] = dw0;
  g1[1] = (int)((tdim0 & 0xFFFFu) << 16);                      // tensor_dim0 lo
  g1[2] = (int)((tdim0 >> 16) | ((tdim1 & 0xFFFFu) << 16));    // dim0 hi | dim1 lo
  g1[3] = (int)((tdim1 >> 16) | (tile0 << 16));                // dim1 hi | tile_dim0
  g1[4] = (int)(tile1 & 0xFFFFu);                              // tile_dim1 (0 = 1D)
  g1[5] = (int)stride0;                                        // tensor_dim0_stride lo
  g1[6] = 0;
  g1[7] = 0;
  i32x4 z4; z4[0]=0; z4[1]=0; z4[2]=0; z4[3]=0;
  i32x8 z8; z8[0]=0; z8[1]=0; z8[2]=0; z8[3]=0; z8[4]=0; z8[5]=0; z8[6]=0; z8[7]=0;
  __builtin_amdgcn_tensor_load_to_lds(g0, g1, z4, z4, z8, 0);
}
#endif

// ---------------------------------------------------------------------------
// Kernel 1: fused Q/K/V projection.  X[8192,1024]f32 @ W[1024,1024]f32 + bias
// -> f16 head-split buffers.  z=0: Qh (scaled by 1/8), z=1: Kh, z=2: Vt
// (V stored transposed [bh][d][l] so attention can TDM it straight to LDS).
// Tiles: 128x128, 8 waves, K-chunk 32. All staging fixed-trip + vectorized.
// ---------------------------------------------------------------------------
__global__ __launch_bounds__(256) void qkv_proj_kernel(
    const float* __restrict__ xq, const float* __restrict__ xk, const float* __restrict__ xv,
    const float* __restrict__ wq, const float* __restrict__ wk, const float* __restrict__ wv,
    const float* __restrict__ bq, const float* __restrict__ bk, const float* __restrict__ bv,
    half_t* __restrict__ Qh, half_t* __restrict__ Kh, half_t* __restrict__ Vt)
{
  __shared__ _Float16 lds_x[128 * 40];   // padded rows: 32 halves + 8 pad
  __shared__ _Float16 lds_w[128 * 40];   // W^T tile: [n][k]
  const int z  = blockIdx.z;
  const float* X  = (z == 0) ? xq : (z == 1) ? xk : xv;
  const float* W  = (z == 0) ? wq : (z == 1) ? wk : wv;
  const float* Bi = (z == 0) ? bq : (z == 1) ? bk : bv;
  const int m0 = blockIdx.x * 128;
  const int n0 = blockIdx.y * 128;
  const int tid  = threadIdx.x;
  const int wave = tid >> 5, lane = tid & 31;
  const int l16  = lane & 15, lhi = (lane >> 4) & 1;
  const int kb0  = lhi * 8, kb1 = 16 + lhi * 8;

  v8f acc[8];
  const v8f vzero = {0.f,0.f,0.f,0.f,0.f,0.f,0.f,0.f};
#pragma unroll
  for (int i = 0; i < 8; ++i) acc[i] = vzero;

  for (int k0 = 0; k0 < 1024; k0 += 32) {
    // X tile: 128 rows x 32 k, f32 -> f16, 4-wide (float4 load, b64 LDS store)
#pragma unroll
    for (int i = 0; i < 4; ++i) {
      int e = tid + i * 256;            // [0,1024), 4 elems each
      int r = e >> 3;
      int c = (e & 7) * 4;
      float4 f = *(const float4*)&X[(size_t)(m0 + r) * 1024 + k0 + c];
      v4h hv; hv[0] = (_Float16)f.x; hv[1] = (_Float16)f.y;
              hv[2] = (_Float16)f.z; hv[3] = (_Float16)f.w;
      *(v4h*)&lds_x[r * 40 + c] = hv;
    }
    // W tile transposed into [n][k]: coalesced float4 along n, 4 b16 stores
#pragma unroll
    for (int i = 0; i < 4; ++i) {
      int e = tid + i * 256;            // [0,1024)
      int n = (e & 31) * 4;
      int c = e >> 5;                   // k 0..31
      float4 f = *(const float4*)&W[(size_t)(k0 + c) * 1024 + n0 + n];
      lds_w[(n + 0) * 40 + c] = (_Float16)f.x;
      lds_w[(n + 1) * 40 + c] = (_Float16)f.y;
      lds_w[(n + 2) * 40 + c] = (_Float16)f.z;
      lds_w[(n + 3) * 40 + c] = (_Float16)f.w;
    }
    __syncthreads();
    const _Float16* xr = &lds_x[(wave * 16 + l16) * 40];
    v16h a = cat16(ld8(xr + kb0), ld8(xr + kb1));
    v16h bf[8];
#pragma unroll
    for (int nt = 0; nt < 8; ++nt) {
      const _Float16* wr = &lds_w[(nt * 16 + l16) * 40];
      bf[nt] = cat16(ld8(wr + kb0), ld8(wr + kb1));
    }
#pragma unroll
    for (int nt = 0; nt < 8; ++nt)
      acc[nt] = wmma16(a, bf[nt], acc[nt]);
    __syncthreads();
  }

#pragma unroll
  for (int nt = 0; nt < 8; ++nt) {
#pragma unroll
    for (int r = 0; r < 8; ++r) {
      int mg = m0 + wave * 16 + r + lhi * 8;
      int ng = n0 + nt * 16 + l16;
      float v = acc[nt][r] + Bi[ng];
      int b = mg >> 11, l = mg & 2047, h = ng >> 6, d = ng & 63;
      size_t hd = ((size_t)b * 16 + h);
      if (z == 0)      Qh[(hd * 2048 + l) * 64 + d] = (_Float16)(v * 0.125f);
      else if (z == 1) Kh[(hd * 2048 + l) * 64 + d] = (_Float16)v;
      else             Vt[(hd * 64 + d) * 2048 + l] = (_Float16)v;
    }
  }
}

// ---------------------------------------------------------------------------
// Kernel 2: attention. One WG = one (b,h) x 128 q-rows; wave owns 16 rows.
// Pass 1: flash running max/sum over K (WMMA QK^T).  Pass 2: recompute S,
// normalize, single write of attn, accumulate O = P@V via WMMA.
// K/V via TDM (padded LDS rows); mask tile staged coalesced through LDS.
// ---------------------------------------------------------------------------
__global__ __launch_bounds__(256) void attn_kernel(
    const half_t* __restrict__ Qh, const half_t* __restrict__ Kh,
    const half_t* __restrict__ Vt, const unsigned char* __restrict__ mask,
    float* __restrict__ attn_out, half_t* __restrict__ Ob)
{
  __shared__ _Float16 lds_k[32 * 72];      // K chunk  [n_local][d] rows padded to 72
  __shared__ _Float16 lds_v[64 * 40];      // V chunk  [d][n_local] rows padded to 40
  __shared__ _Float16 lds_p[8 * 16 * 40];  // per-wave P tile for layout swap
  __shared__ __align__(16) unsigned char lds_m[128 * 32];  // mask tile

  const int bh = blockIdx.y;
  const int b  = bh >> 4, h = bh & 15;
  const int q0 = blockIdx.x * 128;
  const int tid  = threadIdx.x;
  const int wave = tid >> 5, lane = tid & 31;
  const int l16  = lane & 15, lhi = (lane >> 4) & 1;
  const int kb0  = lhi * 8, kb1 = 16 + lhi * 8;
  const int mrow_l = wave * 16;            // wave's row base inside the 128-tile

  // Q fragments (A-layout) straight from global, once.
  const half_t* qp = Qh + ((size_t)bh * 2048 + (q0 + mrow_l + l16)) * 64;
  v16h qa0 = cat16(ld8(qp + kb0),      ld8(qp + kb1));
  v16h qa1 = cat16(ld8(qp + 32 + kb0), ld8(qp + 32 + kb1));

  // mask tile staging base: one coalesced uint4 (16B) per thread per chunk
  const int mr = tid >> 1;                 // row 0..127
  const int mc = (tid & 1) * 16;           // 0 or 16

  float rmax[8], rsum[8];
#pragma unroll
  for (int r = 0; r < 8; ++r) { rmax[r] = -3.0e38f; rsum[r] = 0.0f; }

  // ---------------- pass 1: row max / sum ----------------
  for (int kk = 0; kk < 2048; kk += 32) {
#ifdef USE_TDM
    if (tid < 32) {
      tdm_load((uint32_t)(uintptr_t)&lds_k[0],
               Kh + ((size_t)bh * 2048 + kk) * 64,
               TDM_DW0_K, 2048, 1, 2048, 0, 2048);       // 1D 4KB, padded rows
      __builtin_amdgcn_s_wait_tensorcnt(0);
    }
#else
    {
      const half_t* gk = Kh + ((size_t)bh * 2048 + kk) * 64;
#pragma unroll
      for (int i = 0; i < 1; ++i) {
        int e = tid;                       // 2048 halfs / 256 = 8 halfs each
        int r = e >> 3, c = (e & 7) * 8;
        *(v8h*)&lds_k[r * 72 + c] = *(const v8h*)&gk[r * 64 + c];
      }
    }
#endif
    *(uint4*)&lds_m[mr * 32 + mc] =
        *(const uint4*)&mask[((size_t)b * 2048 + q0 + mr) * 2048 + kk + mc];
    if (kk + 32 < 2048)
      __builtin_prefetch(Kh + ((size_t)bh * 2048 + kk + 32) * 64 + lane * 128, 0, 1);
    __syncthreads();
#pragma unroll
    for (int nt = 0; nt < 2; ++nt) {
      const _Float16* kr = &lds_k[(nt * 16 + l16) * 72];
      v16h b0 = cat16(ld8(kr + kb0),      ld8(kr + kb1));
      v16h b1 = cat16(ld8(kr + 32 + kb0), ld8(kr + 32 + kb1));
      v8f s = {0.f,0.f,0.f,0.f,0.f,0.f,0.f,0.f};
      s = wmma16(qa0, b0, s);
      s = wmma16(qa1, b1, s);
#pragma unroll
      for (int r = 0; r < 8; ++r) {
        float val = s[r];
        if (lds_m[(mrow_l + r + lhi * 8) * 32 + nt * 16 + l16]) val = -3.0e38f;
        float mo = rmax[r], mn = fmaxf(mo, val);
        rsum[r] = rsum[r] * __expf(mo - mn) + __expf(val - mn);
        rmax[r] = mn;
      }
    }
    __syncthreads();
  }

  // reduce max/sum across the 16 lanes sharing each row group
#pragma unroll
  for (int r = 0; r < 8; ++r) {
    float m = rmax[r], l = rsum[r];
#pragma unroll
    for (int off = 1; off < 16; off <<= 1) {
      float om = __shfl_xor(m, off, 16);
      float ol = __shfl_xor(l, off, 16);
      float mn = fmaxf(m, om);
      l = l * __expf(m - mn) + ol * __expf(om - mn);
      m = mn;
    }
    rmax[r] = m;
    rsum[r] = 1.0f / l;  // now holds 1/sum
  }

  // ---------------- pass 2: write attn + O = P@V ----------------
  v8f oacc[4];
  const v8f vzero = {0.f,0.f,0.f,0.f,0.f,0.f,0.f,0.f};
#pragma unroll
  for (int t = 0; t < 4; ++t) oacc[t] = vzero;

  _Float16* pw = &lds_p[wave * 16 * 40];

  for (int kk = 0; kk < 2048; kk += 32) {
#ifdef USE_TDM
    if (tid < 32) {
      tdm_load((uint32_t)(uintptr_t)&lds_k[0],
               Kh + ((size_t)bh * 2048 + kk) * 64,
               TDM_DW0_K, 2048, 1, 2048, 0, 2048);
      tdm_load((uint32_t)(uintptr_t)&lds_v[0],
               Vt + (size_t)bh * 64 * 2048 + kk,
               TDM_DW0_V, 32, 64, 32, 64, 2048);         // 2D 64 rows x 64B, padded
      __builtin_amdgcn_s_wait_tensorcnt(0);
    }
#else
    {
      const half_t* gk = Kh + ((size_t)bh * 2048 + kk) * 64;
      {
        int e = tid;
        int r = e >> 3, c = (e & 7) * 8;
        *(v8h*)&lds_k[r * 72 + c] = *(const v8h*)&gk[r * 64 + c];
      }
      const half_t* gv = Vt + (size_t)bh * 64 * 2048 + kk;
#pragma unroll
      for (int i = 0; i < 8; ++i) {
        int e = tid + i * 256;             // 2048 halfs, scalar per d-row
        int d = e >> 5, n = e & 31;
        lds_v[d * 40 + n] = gv[(size_t)d * 2048 + n];
      }
    }
#endif
    *(uint4*)&lds_m[mr * 32 + mc] =
        *(const uint4*)&mask[((size_t)b * 2048 + q0 + mr) * 2048 + kk + mc];
    if (kk + 32 < 2048)
      __builtin_prefetch(Kh + ((size_t)bh * 2048 + kk + 32) * 64 + lane * 128, 0, 1);
    __syncthreads();

#pragma unroll
    for (int nt = 0; nt < 2; ++nt) {
      const _Float16* kr = &lds_k[(nt * 16 + l16) * 72];
      v16h b0 = cat16(ld8(kr + kb0),      ld8(kr + kb1));
      v16h b1 = cat16(ld8(kr + 32 + kb0), ld8(kr + 32 + kb1));
      v8f s = {0.f,0.f,0.f,0.f,0.f,0.f,0.f,0.f};
      s = wmma16(qa0, b0, s);
      s = wmma16(qa1, b1, s);
#pragma unroll
      for (int r = 0; r < 8; ++r) {
        int mg = q0 + mrow_l + r + lhi * 8;
        int ng = kk + nt * 16 + l16;
        float val = s[r];
        if (lds_m[(mrow_l + r + lhi * 8) * 32 + nt * 16 + l16]) val = -3.0e38f;
        float p = __expf(val - rmax[r]) * rsum[r];
        attn_out[((size_t)bh * 2048 + mg) * 2048 + ng] = p;      // single attn write
        pw[(r + lhi * 8) * 40 + nt * 16 + l16] = (_Float16)p;    // C-layout -> LDS
      }
    }
    // wave-local LDS RAW: wait for the P stores before re-reading as A-layout
    asm volatile("s_wait_dscnt 0" ::: "memory");
    const _Float16* pr = &pw[l16 * 40];
    v16h pa = cat16(ld8(pr + kb0), ld8(pr + kb1));
    v16h bv[4];
#pragma unroll
    for (int t = 0; t < 4; ++t) {
      const _Float16* vr = &lds_v[(t * 16 + l16) * 40];
      bv[t] = cat16(ld8(vr + kb0), ld8(vr + kb1));
    }
#pragma unroll
    for (int t = 0; t < 4; ++t)
      oacc[t] = wmma16(pa, bv[t], oacc[t]);
    __syncthreads();
  }

  // write O (already normalized) as f16 in [b, l, h*64+d] GEMM-ready layout
#pragma unroll
  for (int t = 0; t < 4; ++t) {
#pragma unroll
    for (int r = 0; r < 8; ++r) {
      int mg = q0 + mrow_l + r + lhi * 8;
      int d  = t * 16 + l16;
      Ob[((size_t)b * 2048 + mg) * 1024 + h * 64 + d] = (_Float16)oacc[t][r];
    }
  }
}

// ---------------------------------------------------------------------------
// Kernel 3: out = Ob(f16) @ fc_w + fc_b + residual q   (128x128 tiles)
// ---------------------------------------------------------------------------
__global__ __launch_bounds__(256) void out_proj_kernel(
    const half_t* __restrict__ Ob, const float* __restrict__ fcw,
    const float* __restrict__ fcb, const float* __restrict__ qres,
    float* __restrict__ out)
{
  __shared__ _Float16 lds_a[128 * 40];
  __shared__ _Float16 lds_w[128 * 40];
  const int m0 = blockIdx.x * 128;
  const int n0 = blockIdx.y * 128;
  const int tid  = threadIdx.x;
  const int wave = tid >> 5, lane = tid & 31;
  const int l16  = lane & 15, lhi = (lane >> 4) & 1;
  const int kb0  = lhi * 8, kb1 = 16 + lhi * 8;

  v8f acc[8];
  const v8f vzero = {0.f,0.f,0.f,0.f,0.f,0.f,0.f,0.f};
#pragma unroll
  for (int i = 0; i < 8; ++i) acc[i] = vzero;

  for (int k0 = 0; k0 < 1024; k0 += 32) {
    // A tile: f16 global -> LDS, 8 halfs (16B) per thread per iter
#pragma unroll
    for (int i = 0; i < 2; ++i) {
      int e = tid + i * 256;              // [0,512)
      int r = e >> 2;
      int c = (e & 3) * 8;
      *(v8h*)&lds_a[r * 40 + c] = *(const v8h*)&Ob[(size_t)(m0 + r) * 1024 + k0 + c];
    }
    // W tile transposed, coalesced float4 along n
#pragma unroll
    for (int i = 0; i < 4; ++i) {
      int e = tid + i * 256;
      int n = (e & 31) * 4;
      int c = e >> 5;
      float4 f = *(const float4*)&fcw[(size_t)(k0 + c) * 1024 + n0 + n];
      lds_w[(n + 0) * 40 + c] = (_Float16)f.x;
      lds_w[(n + 1) * 40 + c] = (_Float16)f.y;
      lds_w[(n + 2) * 40 + c] = (_Float16)f.z;
      lds_w[(n + 3) * 40 + c] = (_Float16)f.w;
    }
    __syncthreads();
    const _Float16* ar = &lds_a[(wave * 16 + l16) * 40];
    v16h a = cat16(ld8(ar + kb0), ld8(ar + kb1));
    v16h bf[8];
#pragma unroll
    for (int nt = 0; nt < 8; ++nt) {
      const _Float16* wr = &lds_w[(nt * 16 + l16) * 40];
      bf[nt] = cat16(ld8(wr + kb0), ld8(wr + kb1));
    }
#pragma unroll
    for (int nt = 0; nt < 8; ++nt)
      acc[nt] = wmma16(a, bf[nt], acc[nt]);
    __syncthreads();
  }

#pragma unroll
  for (int nt = 0; nt < 8; ++nt) {
#pragma unroll
    for (int r = 0; r < 8; ++r) {
      int mg = m0 + wave * 16 + r + lhi * 8;
      int ng = n0 + nt * 16 + l16;
      size_t o = (size_t)mg * 1024 + ng;
      out[o] = acc[nt][r] + fcb[ng] + qres[o];
    }
  }
}

// ---------------------------------------------------------------------------
// Launch
// ---------------------------------------------------------------------------
extern "C" void kernel_launch(void* const* d_in, const int* in_sizes, int n_in,
                              void* d_out, int out_size, void* d_ws, size_t ws_size,
                              hipStream_t stream) {
  (void)in_sizes; (void)n_in; (void)out_size; (void)ws_size;
  const float* q    = (const float*)d_in[0];
  const float* k    = (const float*)d_in[1];
  const float* v    = (const float*)d_in[2];
  const unsigned char* mask = (const unsigned char*)d_in[3];
  const float* w_qs = (const float*)d_in[4];
  const float* b_qs = (const float*)d_in[5];
  const float* w_ks = (const float*)d_in[6];
  const float* b_ks = (const float*)d_in[7];
  const float* w_vs = (const float*)d_in[8];
  const float* b_vs = (const float*)d_in[9];
  const float* fc_w = (const float*)d_in[10];
  const float* fc_b = (const float*)d_in[11];

  char* ws = (char*)d_ws;
  const size_t SEG = (size_t)16 * 1024 * 1024;   // 8,388,608 f16 each
  half_t* Qh = (half_t*)(ws);
  half_t* Kh = (half_t*)(ws + SEG);
  half_t* Vt = (half_t*)(ws + 2 * SEG);
  half_t* Ob = (half_t*)(ws + 3 * SEG);

  float* out  = (float*)d_out;
  float* attn = out + (size_t)8388608;           // out first, then attn

  qkv_proj_kernel<<<dim3(64, 8, 3), 256, 0, stream>>>(
      q, k, v, w_qs, w_ks, w_vs, b_qs, b_ks, b_vs, Qh, Kh, Vt);
  attn_kernel<<<dim3(16, 64), 256, 0, stream>>>(Qh, Kh, Vt, mask, attn, Ob);
  out_proj_kernel<<<dim3(64, 8), 256, 0, stream>>>(Ob, fc_w, fc_b, q, out);
}
